// RemizovShiftLocal_46248207843817
// MI455X (gfx1250) — matrix operand order repaired and checked by reading, not compiled, not run on hardware
//
#include <hip/hip_runtime.h>
#include <math.h>

typedef __bf16 bf16;
typedef bf16  v16bf __attribute__((ext_vector_type(16)));
typedef float v8f   __attribute__((ext_vector_type(8)));
typedef unsigned int u32x4 __attribute__((ext_vector_type(4)));
typedef unsigned int u32x8 __attribute__((ext_vector_type(8)));

#define Hdim 192
#define Wdim 192
#define HWSZ (192*192)
#define Cdim 64
#define SHCH 8
#define TAUF 0.1f

// ---------------------------------------------------------------------------
// Kernel 1: h = gelu(conv3x3_SAME(x, w1))  -> ws as (B, 8, H, W) fp32
// Implicit GEMM via v_wmma_f32_16x16x32_bf16, fully unrolled (18 WMMAs).
// Block: 256 thr (8 waves). Tile: 16 (x) x 8 (y) pixels. Wave w owns row y0+w.
// K order: K = tap*64 + c  => each 32-wide k-tile is (one tap, 32 channels).
// ---------------------------------------------------------------------------
__global__ __launch_bounds__(256)
void remizov_conv_wmma(const float* __restrict__ x,
                       const float* __restrict__ w1,
                       float* __restrict__ hbuf) {
  __shared__ bf16 sx[64][10][18];   // [c][y0-1..y0+8][x0-1..x0+16], 23 KB
  __shared__ bf16 sw[16][64][9];    // [n][c][tap]; rows 8..15 zero (N padding), 18 KB
  const int x0  = blockIdx.x * 16;
  const int y0  = blockIdx.y * 8;
  const int b   = blockIdx.z;
  const int tid = threadIdx.x;

  // Stage input halo tile (zero-padded SAME borders), fp32 -> bf16.
  for (int i = tid; i < 64 * 10 * 18; i += 256) {
    int c = i / 180, rem = i % 180;
    int yy = rem / 18, xx = rem % 18;
    int gy = y0 - 1 + yy, gx = x0 - 1 + xx;
    float v = 0.0f;
    if (gy >= 0 && gy < Hdim && gx >= 0 && gx < Wdim)
      v = x[(((size_t)b * Cdim + c) * Hdim + gy) * Wdim + gx];
    sx[c][yy][xx] = (bf16)v;
  }
  // Stage weights (flat copy: n*576 + c*9 + tap matches w1 [SH][C][3][3]);
  // zero-fill the 8 padding output channels so B loads need no predication.
  for (int i = tid; i < 16 * 64 * 9; i += 256)
    (&sw[0][0][0])[i] = (i < SHCH * 64 * 9) ? (bf16)w1[i] : (bf16)0.0f;
  __syncthreads();

  const int wv  = tid >> 5;   // wave id = local output row
  const int ln  = tid & 31;
  const int M   = ln & 15;    // output pixel within row (A-matrix row)
  const int hiK = ln >> 4;    // lane group selects K sub-blocks
  const int N   = ln & 15;    // B/C/D column (shader channel, 8..15 padded)

  // Per-lane invariant LDS bases (half-element indices).
  const bf16* sxp = &sx[0][0][0] + (wv * 18 + M);       // + c*180 + ty*18 + tx
  const bf16* swp = &sw[0][0][0] + (N * 576 + hiK * 144); // + (cbase+i)*9 + tap

  v8f acc = {0.f, 0.f, 0.f, 0.f, 0.f, 0.f, 0.f, 0.f};
  #pragma unroll
  for (int t = 0; t < 18; ++t) {
    const int tap   = t >> 1;
    const int cbase = (t & 1) << 5;          // 0 or 32
    const int ty = tap / 3, tx = tap % 3;    // spatial tap offset (-1 folded)
    v16bf af, bfm;
    // A k order used here (same as B): lanes 0-15 -> kk 0..15, lanes 16-31 -> kk 16..31
    #pragma unroll
    for (int i = 0; i < 16; ++i) {
      int kk = hiK * 16 + i;
      af[i] = sxp[(cbase + kk) * 180 + ty * 18 + tx];
    }
    // B k order used here (same as A): lanes 0-15 halves = K 0..15 (N=lane), lanes 16-31 = K 16..31
    #pragma unroll
    for (int i = 0; i < 16; ++i)
      bfm[i] = swp[(cbase + i) * 9 + tap];

    acc = __builtin_amdgcn_wmma_f32_16x16x32_bf16(
        /*neg_a=*/false, af, /*neg_b=*/false, bfm,
        /*c_mod=*/(short)0, acc, /*reuse_a=*/false, /*reuse_b=*/false);
  }

  // C/D layout: lane group hiK -> M = r + 8*hiK in VGPR r; N = ln & 15.
  if (N < SHCH) {
    const int y = y0 + wv;
    float* hp = hbuf + (((size_t)b * SHCH + N) * Hdim + y) * Wdim + x0;
    #pragma unroll
    for (int r = 0; r < 8; ++r) {
      float v = acc[r];
      v = 0.5f * v * (1.0f + erff(v * 0.70710678118f));   // exact gelu
      hp[r + hiK * 8] = v;
    }
  }
}

// ---------------------------------------------------------------------------
// Kernel 2: per (b,c) plane. TDM-stage 147 KB fp32 plane into LDS, compute
// coeffs from h on the fly, 4 bilinear border-clamped samples from LDS.
// ---------------------------------------------------------------------------
__device__ __forceinline__ float bilin_lds(const float* pl, float gx, float gy) {
  float px = fminf(fmaxf((gx + 1.0f) * 0.5f * (float)(Wdim - 1), 0.0f), (float)(Wdim - 1));
  float py = fminf(fmaxf((gy + 1.0f) * 0.5f * (float)(Hdim - 1), 0.0f), (float)(Hdim - 1));
  float fx0 = floorf(px), fy0 = floorf(py);
  float wx = px - fx0, wy = py - fy0;
  int x0i = (int)fx0, y0i = (int)fy0;
  int x1i = (x0i + 1 < Wdim - 1) ? (x0i + 1) : (Wdim - 1);
  int y1i = (y0i + 1 < Hdim - 1) ? (y0i + 1) : (Hdim - 1);
  float v00 = pl[y0i * Wdim + x0i], v01 = pl[y0i * Wdim + x1i];
  float v10 = pl[y1i * Wdim + x0i], v11 = pl[y1i * Wdim + x1i];
  float top = v00 * (1.0f - wx) + v01 * wx;
  float bot = v10 * (1.0f - wx) + v11 * wx;
  return top * (1.0f - wy) + bot * wy;
}

__global__ __launch_bounds__(512)
void remizov_sample(const float* __restrict__ x,
                    const float* __restrict__ hbuf,
                    const float* __restrict__ w2,
                    const float* __restrict__ b2,
                    const float* __restrict__ log_scale,
                    float* __restrict__ out) {
  extern __shared__ float plane[];          // 192*192 fp32 = 147456 B (CDNA5 320KB LDS)
  const int bc = blockIdx.x;
  const int b  = bc >> 6;
  const int c  = bc & 63;
  const float* xp = x + (size_t)bc * HWSZ;

  // --- Tensor Data Mover: 16 waves each DMA a 192x12 row band into LDS ---
  {
    int uw = __builtin_amdgcn_readfirstlane((int)(threadIdx.x >> 5));
    unsigned long long ga = (unsigned long long)(size_t)xp
                          + (unsigned long long)(unsigned)(uw * (12 * Wdim * 4));
    u32x4 g0; u32x8 g1;
    g0[0] = 1u;                                              // count=1, user D#
    g0[1] = (unsigned)(uw * (12 * Wdim * 4));                // lds_addr (bytes)
    g0[2] = (unsigned)(ga & 0xffffffffull);                  // global_addr lo
    g0[3] = (unsigned)((ga >> 32) & 0x1ffffffull) | (2u << 30); // addr hi | type=2
    g1[0] = (2u << 16);            // data_size = 2 (4 bytes)
    g1[1] = ((unsigned)Wdim) << 16;// tensor_dim0[15:0]=192
    g1[2] = (12u << 16);           // tensor_dim0 hi=0 | tensor_dim1 lo=12
    g1[3] = ((unsigned)Wdim) << 16;// tensor_dim1 hi=0 | tile_dim0=192
    g1[4] = 12u;                   // tile_dim1=12 | tile_dim2=0
    g1[5] = (unsigned)Wdim;        // tensor_dim0_stride lo = 192 elements
    g1[6] = 0u;
    g1[7] = 0u;
    asm volatile("tensor_load_to_lds %0, %1" :: "s"(g0), "s"(g1) : "memory");
    __builtin_amdgcn_s_wait_tensorcnt(0);
  }
  __syncthreads();

  // Per-block constants: w2 rows 4c..4c+3 (K=8 each) and b2, scale.
  float wv[4][8], bb[4];
  #pragma unroll
  for (int k = 0; k < 4; ++k) {
    bb[k] = b2[c * 4 + k];
    #pragma unroll
    for (int j = 0; j < 8; ++j)
      wv[k][j] = w2[(size_t)(c * 4 + k) * 8 + j];
  }
  const float esc = expf(log_scale[0]);
  const float* hb = hbuf + (size_t)b * SHCH * HWSZ;
  float* op = out + (size_t)bc * HWSZ;

  for (int p = threadIdx.x; p < HWSZ; p += 512) {
    const int py = p / Wdim, px = p % Wdim;

    float hv[8];
    #pragma unroll
    for (int j = 0; j < 8; ++j)
      hv[j] = hb[(size_t)j * HWSZ + p];
    __builtin_prefetch(hb + p + 512, 0, 1);   // global_prefetch_b8 for next iter

    float co[4];
    #pragma unroll
    for (int k = 0; k < 4; ++k) {
      float acc = bb[k];
      #pragma unroll
      for (int j = 0; j < 8; ++j) acc = fmaf(hv[j], wv[k][j], acc);
      co[k] = acc;
    }

    // stable softplus
    float a  = fmaxf(co[0], 0.0f) + log1pf(expf(-fabsf(co[0])));
    float s  = sqrtf(a * TAUF + 1e-8f);
    float dx = co[1] * TAUF;
    float dy = co[2] * TAUF;
    float cc = fminf(fmaxf(co[3], -5.0f), 5.0f);

    const float gx = (float)px * (2.0f / (float)(Wdim - 1)) - 1.0f;
    const float gy = (float)py * (2.0f / (float)(Hdim - 1)) - 1.0f;

    float sum = bilin_lds(plane, gx + s + dx, gy + dy)
              + bilin_lds(plane, gx - s + dx, gy + dy)
              + bilin_lds(plane, gx + dx,     gy + s + dy)
              + bilin_lds(plane, gx + dx,     gy - s + dy);

    float xv = plane[p];
    op[p] = esc * (0.25f * sum + TAUF * cc * xv);
  }
}

// ---------------------------------------------------------------------------
extern "C" void kernel_launch(void* const* d_in, const int* in_sizes, int n_in,
                              void* d_out, int out_size, void* d_ws, size_t ws_size,
                              hipStream_t stream) {
  const float* x   = (const float*)d_in[0];
  const float* w1  = (const float*)d_in[1];
  const float* w2  = (const float*)d_in[2];
  const float* b2  = (const float*)d_in[3];
  const float* ls  = (const float*)d_in[4];
  float* out  = (float*)d_out;
  float* hbuf = (float*)d_ws;          // (8, 8, 192, 192) fp32 = 9.44 MB

  dim3 gridc(Wdim / 16, Hdim / 8, 8);  // (12, 24, 8)
  remizov_conv_wmma<<<gridc, 256, 0, stream>>>(x, w1, hbuf);

  remizov_sample<<<dim3(8 * Cdim), 512, HWSZ * sizeof(float), stream>>>(
      x, hbuf, w2, b2, ls, out);
}
